// MultiTaskGATv2_53360673686084
// MI455X (gfx1250) — matrix operand order repaired and checked by reading, not compiled, not run on hardware
//
#include <hip/hip_runtime.h>
#include <math.h>

typedef __attribute__((ext_vector_type(16))) __bf16 v16bf;
typedef __attribute__((ext_vector_type(8)))  float  v8f;

namespace {
constexpr int NN    = 50000;
constexpr int EE    = 200000;
constexpr int HID   = 128;
constexpr int HEADS = 4;
constexpr int NLAY  = 3;
constexpr int NCLS  = 9;
constexpr int ECLS  = 6;
}

__device__ __forceinline__ float gelu_f(float v) {
  return 0.5f * v * (1.0f + erff(v * 0.70710678118654752440f));
}
__device__ __forceinline__ unsigned short f2bf(float x) {
  unsigned u = __float_as_uint(x);
  u = (u + 0x7FFFu + ((u >> 16) & 1u)) >> 16;   // RNE fp32 -> bf16
  return (unsigned short)u;
}
__device__ __forceinline__ unsigned pk2bf(float a, float b) {
  return (unsigned)f2bf(a) | ((unsigned)f2bf(b) << 16);
}
// monotone float<->uint encoding so unsigned atomicMax == float max
__device__ __forceinline__ unsigned encf(float f) {
  unsigned u = __float_as_uint(f);
  return (u & 0x80000000u) ? ~u : (u | 0x80000000u);
}
__device__ __forceinline__ float decf(unsigned u) {
  return __uint_as_float((u & 0x80000000u) ? (u & 0x7FFFFFFFu) : ~u);
}
// wave-relative LDS byte offset of a __shared__ object (flat addr low 32 bits)
__device__ __forceinline__ unsigned lds_off(const void* p) {
  return (unsigned)(unsigned long long)p;
}

union FragBF { unsigned u[8]; v16bf v; };

// ---------------------------------------------------------------------------
// WMMA GEMM: C[M,128] = act(A[M,K] @ B[K,128] (+bias) (+Cprev))
// A rows optionally gathered via gidx. A tile is DMA'd fp32 into LDS with
// GLOBAL_LOAD_ASYNC_TO_LDS_B128 (ASYNCcnt), converted to bf16 at fragment
// build; B tile is converted+transposed to bf16 during staging.
// K must be a multiple of 32; N fixed at 128.
// Block: 256 threads = 8 waves, 128 rows per block, 16 rows per wave.
// ---------------------------------------------------------------------------
template<int ACT, bool BIAS, bool ACC, bool GATHER>
__global__ __launch_bounds__(256) void k_gemm_wmma(
    const float* __restrict__ A, const float* __restrict__ B,
    const float* __restrict__ bias, float* __restrict__ C,
    const int* __restrict__ gidx, int M, int K)
{
  __shared__ __attribute__((aligned(16))) float          lAf[128][32]; // fp32 A tile (async dest)
  __shared__ __attribute__((aligned(16))) unsigned short lBs[128][32]; // bf16 B tile, transposed [n][k]

  const int t    = threadIdx.x;
  const int wave = t >> 5;
  const int lane = t & 31;
  const int m0   = blockIdx.x * 128;

  v8f acc[8] = {};

  const int nk = K >> 5;
  for (int kt = 0; kt < nk; ++kt) {
    __syncthreads();
    // ---- stage A tile: async DMA global(fp32) -> LDS, 16B per lane x4 ----
#pragma unroll
    for (int i = 0; i < 4; ++i) {
      const int b  = t + i * 256;        // 0..1023 quad id
      const int r  = b >> 3;             // row 0..127
      const int k4 = (b & 7) << 2;       // k offset 0,4,..,28
      const int gm = m0 + r;
      const unsigned lds = lds_off(&lAf[r][k4]);
      if (gm < M) {
        const long row = GATHER ? (long)gidx[gm] : (long)gm;
        const float* gp = A + row * (long)K + (kt << 5) + k4;
        asm volatile("global_load_async_to_lds_b128 %0, %1, off"
                     :: "v"(lds), "v"(gp) : "memory");
      } else {
        float4 z = {0.f, 0.f, 0.f, 0.f};
        *(float4*)&lAf[r][k4] = z;       // zero-fill OOB rows
      }
    }
    // ---- stage B tile transposed: read B[k][n0..n0+16), write lBs[n][k] ----
    {
      const int kk = t >> 3;
      const int n0 = (t & 7) << 4;
      const float* sp = B + (long)((kt << 5) + kk) * 128 + n0;
      if (kt + 1 < nk) __builtin_prefetch(sp + 32 * 128, 0, 0);
#pragma unroll
      for (int i = 0; i < 4; ++i) {
        float4 v4 = ((const float4*)sp)[i];
        lBs[n0 + i*4 + 0][kk] = f2bf(v4.x);
        lBs[n0 + i*4 + 1][kk] = f2bf(v4.y);
        lBs[n0 + i*4 + 2][kk] = f2bf(v4.z);
        lBs[n0 + i*4 + 3][kk] = f2bf(v4.w);
      }
    }
    asm volatile("s_wait_asynccnt 0x0" ::: "memory");  // wave's DMA done
    __syncthreads();                                   // all waves' tiles visible

    // ---- A fragment (ISA 7.12.2: 16-bit A 16x32), fp32->bf16 pack ----
    FragBF af;
    {
      const float* ap = &lAf[wave * 16 + (lane & 15)][0];
      const int kb = (lane >> 4) << 3;     // k base 0 or 8
#pragma unroll
      for (int v = 0; v < 4; ++v) {
        float2 p0 = *(const float2*)&ap[kb + 2*v];        // k = base + 0..7
        float2 p1 = *(const float2*)&ap[16 + kb + 2*v];   // k = 16 + base + 0..7
        af.u[v]     = pk2bf(p0.x, p0.y);
        af.u[v + 4] = pk2bf(p1.x, p1.y);
      }
    }
    // ---- preload all 8 B fragments, then back-to-back WMMA ----
    FragBF bf[8];
#pragma unroll
    for (int nt = 0; nt < 8; ++nt) {
      const unsigned* bp = (const unsigned*)&lBs[nt * 16 + (lane & 15)][0];
      const int ks2 = (lane >> 4) << 2;    // k-pair base 0 or 4 (k=0/8), +8 pairs for upper half
#pragma unroll
      for (int v = 0; v < 8; ++v) bf[nt].u[v] = bp[ks2 + v + (v >= 4 ? 4 : 0)];
    }
#pragma unroll
    for (int nt = 0; nt < 8; ++nt) {
      acc[nt] = __builtin_amdgcn_wmma_f32_16x16x32_bf16(
          false, af.v, false, bf[nt].v, (short)0, acc[nt], false, false);
    }
  }

  // ---- epilogue: C layout M = r + 8*(lane>=16), N = lane%16 ----
  const int rbase = m0 + wave * 16 + ((lane >> 4) << 3);
  const int nlow  = lane & 15;
#pragma unroll
  for (int nt = 0; nt < 8; ++nt) {
    const int col = nt * 16 + nlow;
    const float bv = BIAS ? bias[col] : 0.f;
#pragma unroll
    for (int r = 0; r < 8; ++r) {
      const int row = rbase + r;
      if (row < M) {
        float v = acc[nt][r];
        if (ACC) v += C[(long)row * 128 + col];
        v += bv;
        if (ACT == 1) v = gelu_f(v);
        C[(long)row * 128 + col] = v;
      }
    }
  }
}

// ---------------------------------------------------------------------------
// utility / graph kernels
// ---------------------------------------------------------------------------
__global__ void k_fill_u32(unsigned* __restrict__ p, unsigned v, int n) {
  int i = blockIdx.x * 256 + threadIdx.x;
  if (i < n) p[i] = v;
}

// self-loop attr: sum incoming ea per dst + count
__global__ void k_loop_accum(const int* __restrict__ dst, const float* __restrict__ ea,
                             float* __restrict__ lsum, float* __restrict__ cnt, int E) {
  int e = blockIdx.x * 256 + threadIdx.x;
  if (e >= E) return;
  int d = dst[e];
  atomicAdd(&cnt[d], 1.0f);
  const float* row = ea + (long)e * HID;
  float* out = lsum + (long)d * HID;
  for (int c = 0; c < HID; ++c) atomicAdd(&out[c], row[c]);
}
__global__ void k_loop_div(float* __restrict__ lsum, const float* __restrict__ cnt, int n) {
  int i = blockIdx.x * 256 + threadIdx.x;
  if (i < n) lsum[i] /= fmaxf(cnt[i >> 7], 1.0f);
}

// pass A: logits[e][h] = sum_c leaky(xl[s]+xr[d]+ee[e]) * att ; segment max
__global__ void k_gat_logits(const int* __restrict__ src, const int* __restrict__ dst,
                             const float* __restrict__ xl, const float* __restrict__ xr,
                             const float* __restrict__ ee, const float* __restrict__ att,
                             float* __restrict__ logit, unsigned* __restrict__ mx,
                             int E, int N) {
  int e = blockIdx.x * 256 + threadIdx.x;
  if (e >= E + N) return;
  int s, d;
  if (e < E) { s = src[e]; d = dst[e]; } else { s = d = e - E; }
  const float* pl = xl + (long)s * HID;
  const float* pr = xr + (long)d * HID;
  const float* pe = ee + (long)e * HID;
  for (int h = 0; h < HEADS; ++h) {
    float acc = 0.f;
#pragma unroll
    for (int c = 0; c < 32; ++c) {
      float m = pl[h*32+c] + pr[h*32+c] + pe[h*32+c];
      m = (m > 0.f) ? m : 0.2f * m;            // leaky_relu, slope 0.2
      acc += m * att[h*32+c];
    }
    logit[(long)e * HEADS + h] = acc;
    atomicMax(&mx[(long)d * HEADS + h], encf(acc));
  }
}

// pass B: p = exp(logit - max); segment sum
__global__ void k_gat_expsum(const int* __restrict__ dst, float* __restrict__ logit,
                             const unsigned* __restrict__ mx, float* __restrict__ sums,
                             int E, int N) {
  int e = blockIdx.x * 256 + threadIdx.x;
  if (e >= E + N) return;
  int d = (e < E) ? dst[e] : (e - E);
  for (int h = 0; h < HEADS; ++h) {
    float p = expf(logit[(long)e * HEADS + h] - decf(mx[(long)d * HEADS + h]));
    logit[(long)e * HEADS + h] = p;
    atomicAdd(&sums[(long)d * HEADS + h], p);
  }
}

// pass C: out[d] += xl[s] * alpha
__global__ void k_gat_scatter(const int* __restrict__ src, const int* __restrict__ dst,
                              const float* __restrict__ xl, const float* __restrict__ logit,
                              const float* __restrict__ sums, float* __restrict__ out,
                              int E, int N) {
  int e = blockIdx.x * 256 + threadIdx.x;
  if (e >= E + N) return;
  int s, d;
  if (e < E) { s = src[e]; d = dst[e]; } else { s = d = e - E; }
  const float* pl = xl + (long)s * HID;
  float* po = out + (long)d * HID;
  for (int h = 0; h < HEADS; ++h) {
    float alpha = logit[(long)e * HEADS + h] / (sums[(long)d * HEADS + h] + 1e-16f);
#pragma unroll
    for (int c = 0; c < 32; ++c) atomicAdd(&po[h*32+c], pl[h*32+c] * alpha);
  }
}

// pass D: h = gelu(out + conv_bias) + h   (residual)
__global__ void k_gat_update(float* __restrict__ h, const float* __restrict__ out,
                             const float* __restrict__ cb, int n) {
  int i = blockIdx.x * 256 + threadIdx.x;
  if (i < n) h[i] = gelu_f(out[i] + cb[i & 127]) + h[i];
}

// small head: out[r][o] = hid[r] . w2[:,o] + b2[o]
template<int OUT>
__global__ void k_head(const float* __restrict__ hid, const float* __restrict__ w2,
                       const float* __restrict__ b2, float* __restrict__ out, int M) {
  int r = blockIdx.x * 256 + threadIdx.x;
  if (r >= M) return;
  float a[OUT];
#pragma unroll
  for (int o = 0; o < OUT; ++o) a[o] = b2[o];
  const float* hr = hid + (long)r * HID;
  for (int c = 0; c < HID; ++c) {
    float hv = hr[c];
    const float* wr = w2 + c * OUT;
#pragma unroll
    for (int o = 0; o < OUT; ++o) a[o] += hv * wr[o];
  }
#pragma unroll
  for (int o = 0; o < OUT; ++o) out[(long)r * OUT + o] = a[o];
}

// ---------------------------------------------------------------------------
extern "C" void kernel_launch(void* const* d_in, const int* in_sizes, int n_in,
                              void* d_out, int out_size, void* d_ws, size_t ws_size,
                              hipStream_t stream)
{
  (void)in_sizes; (void)n_in; (void)out_size; (void)ws_size;

  const float* x       = (const float*)d_in[0];
  const float* eattr   = (const float*)d_in[1];
  const int*   eidx    = (const int*)  d_in[2];
  const float* node_w  = (const float*)d_in[3];
  const float* node_b  = (const float*)d_in[4];
  const float* edge_w  = (const float*)d_in[5];
  const float* edge_b  = (const float*)d_in[6];
  const float* lin_l_w = (const float*)d_in[7];
  const float* lin_l_b = (const float*)d_in[8];
  const float* lin_r_w = (const float*)d_in[9];
  const float* lin_r_b = (const float*)d_in[10];
  const float* lin_e_w = (const float*)d_in[11];
  const float* att     = (const float*)d_in[12];
  const float* conv_b  = (const float*)d_in[13];
  const float* nc_w1   = (const float*)d_in[14];
  const float* nc_b1   = (const float*)d_in[15];
  const float* nc_w2   = (const float*)d_in[16];
  const float* nc_b2   = (const float*)d_in[17];
  const float* ec_w1   = (const float*)d_in[18];
  const float* ec_b1   = (const float*)d_in[19];
  const float* ec_w2   = (const float*)d_in[20];
  const float* ec_b2   = (const float*)d_in[21];

  const int* src = eidx;        // edge_index[0]
  const int* dst = eidx + EE;   // edge_index[1]

  float* ws = (float*)d_ws;
  size_t o = 0;
  float* h     = ws + o; o += (size_t)NN * HID;
  float* ea    = ws + o; o += (size_t)EE * HID;
  float* lattr = ws + o; o += (size_t)NN * HID;
  float* ee    = ws + o; o += (size_t)(EE + NN) * HID;
  float* xl    = ws + o; o += (size_t)NN * HID;
  float* xr    = ws + o; o += (size_t)NN * HID;
  float* oacc  = ws + o; o += (size_t)NN * HID;
  float* logit = ws + o; o += (size_t)(EE + NN) * HEADS;
  float* sums  = ws + o; o += (size_t)NN * HEADS;
  unsigned* mx = (unsigned*)(ws + o); o += (size_t)NN * HEADS;
  float* cnt   = ws + o; o += (size_t)NN;
  float* out_f = (float*)d_out;
  float* hid_n = xl;   // reuse after GAT layers
  float* hid_e = ee;   // reuse after GAT layers

  const int T = EE + NN;
  auto gblk = [](int M) { return dim3((unsigned)((M + 127) / 128)); };
  auto eblk = [](int n) { return dim3((unsigned)((n + 255) / 256)); };

  // initial projections (K=768), gelu+bias
  k_gemm_wmma<1, true,  false, false><<<gblk(NN), 256, 0, stream>>>(x,     node_w, node_b, h,  nullptr, NN, 768);
  k_gemm_wmma<1, true,  false, false><<<gblk(EE), 256, 0, stream>>>(eattr, edge_w, edge_b, ea, nullptr, EE, 768);

  // self-loop edge attr = mean of incoming ea  (fill_value='mean')
  k_fill_u32<<<eblk(NN * HID), 256, 0, stream>>>((unsigned*)lattr, 0u, NN * HID);
  k_fill_u32<<<eblk(NN),       256, 0, stream>>>((unsigned*)cnt,   0u, NN);
  k_loop_accum<<<eblk(EE), 256, 0, stream>>>(dst, ea, lattr, cnt, EE);
  k_loop_div  <<<eblk(NN * HID), 256, 0, stream>>>(lattr, cnt, NN * HID);

  for (int l = 0; l < NLAY; ++l) {
    const float* wl = lin_l_w + (size_t)l * HID * HID;
    const float* wr = lin_r_w + (size_t)l * HID * HID;
    const float* we = lin_e_w + (size_t)l * HID * HID;
    k_gemm_wmma<0, true,  false, false><<<gblk(NN), 256, 0, stream>>>(h,     wl, lin_l_b + l * HID, xl, nullptr, NN, HID);
    k_gemm_wmma<0, true,  false, false><<<gblk(NN), 256, 0, stream>>>(h,     wr, lin_r_b + l * HID, xr, nullptr, NN, HID);
    k_gemm_wmma<0, false, false, false><<<gblk(EE), 256, 0, stream>>>(ea,    we, nullptr, ee,                    nullptr, EE, HID);
    k_gemm_wmma<0, false, false, false><<<gblk(NN), 256, 0, stream>>>(lattr, we, nullptr, ee + (size_t)EE * HID, nullptr, NN, HID);

    k_fill_u32<<<eblk(NN * HEADS), 256, 0, stream>>>(mx,              0u, NN * HEADS);
    k_fill_u32<<<eblk(NN * HEADS), 256, 0, stream>>>((unsigned*)sums, 0u, NN * HEADS);
    k_fill_u32<<<eblk(NN * HID),   256, 0, stream>>>((unsigned*)oacc, 0u, NN * HID);

    k_gat_logits <<<eblk(T), 256, 0, stream>>>(src, dst, xl, xr, ee, att + (size_t)l * HEADS * 32, logit, mx, EE, NN);
    k_gat_expsum <<<eblk(T), 256, 0, stream>>>(dst, logit, mx, sums, EE, NN);
    k_gat_scatter<<<eblk(T), 256, 0, stream>>>(src, dst, xl, logit, sums, oacc, EE, NN);
    k_gat_update <<<eblk(NN * HID), 256, 0, stream>>>(h, oacc, conv_b + l * HID, NN * HID);
  }

  // node classifier
  k_gemm_wmma<1, true, false, false><<<gblk(NN), 256, 0, stream>>>(h, nc_w1, nc_b1, hid_n, nullptr, NN, HID);
  k_head<NCLS><<<eblk(NN), 256, 0, stream>>>(hid_n, nc_w2, nc_b2, out_f, NN);

  // edge classifier: er = [h[row], h[col], ea] @ ec_w1 done as 3 accumulate GEMMs
  k_gemm_wmma<0, false, false, true ><<<gblk(EE), 256, 0, stream>>>(h,  ec_w1,                  nullptr, hid_e, src, EE, HID);
  k_gemm_wmma<0, false, true,  true ><<<gblk(EE), 256, 0, stream>>>(h,  ec_w1 + 1 * HID * HID,  nullptr, hid_e, dst, EE, HID);
  k_gemm_wmma<1, true,  true,  false><<<gblk(EE), 256, 0, stream>>>(ea, ec_w1 + 2 * HID * HID,  ec_b1,   hid_e, nullptr, EE, HID);
  k_head<ECLS><<<eblk(EE), 256, 0, stream>>>(hid_e, ec_w2, ec_b2, out_f + (size_t)NN * NCLS, EE);
}
